// VQ_VAE_575525617954
// MI455X (gfx1250) — matrix operand order repaired and checked by reading, not compiled, not run on hardware
//
#include <hip/hip_runtime.h>
#include <hip/hip_bf16.h>

// ==========================================================================
// VQ-VAE forward on gfx1250 (MI455X), bf16 WMMA implicit-GEMM pipeline.
// All 256-channel convs -> v_wmma_f32_16x16x32_bf16 (f32 accumulate).
// Each wave computes a 16x64 output tile (4 WMMA accumulators); one A fetch
// feeds 4 WMMAs; invalid conv taps are skipped outright. All mode flags and
// kernel extents are template parameters so the inner loop carries no
// runtime selects (relu = single v_pk_max_i16, no cndmask blending).
// Needs ~110 MB of workspace.
// ==========================================================================

typedef __attribute__((ext_vector_type(16))) __bf16 v16bf;
typedef __attribute__((ext_vector_type(8)))  float  v8f;
typedef __attribute__((ext_vector_type(8)))  short  short8;

union BfPack { short s[16]; v16bf v; };

__device__ __forceinline__ short8 load8(const __bf16* p) {
  return *reinterpret_cast<const short8*>(p);
}
__device__ __forceinline__ short8 relu8(short8 s) {
#pragma unroll
  for (int i = 0; i < 8; ++i) s[i] = (s[i] < 0) ? (short)0 : s[i];
  return s;
}

// --------------------------------------------------------------------------
// Generic WMMA implicit-GEMM conv / convT / matmul kernel.
//   A  : NHWC bf16 activations, Ci fixed = 256
//   W  : bf16 weights, layout [N(co)][K], K = (kh*KW+kw)*256 + ci
//   out: y = (acc + res) * scale[n] + shift[n]   (+optional relu)
// One wave computes a 16x64 output tile. grid = (M/16, N/64), block = 32.
// Output spatial dims are powers of two: Wout = 1<<wShift, Hout = 1<<hShift.
// --------------------------------------------------------------------------
template<int KH, int KW, int CONVT, int RELUIN, int RELUOUT, int F32MODE,
         int HASRES, int HASOUTB>
__global__ __launch_bounds__(32)
void wmma_conv_gemm(const __bf16* __restrict__ A, const __bf16* __restrict__ W,
                    __bf16* __restrict__ outB, float* __restrict__ outF,
                    const __bf16* __restrict__ res,
                    const float* __restrict__ scale, const float* __restrict__ shift,
                    int N, int Hin, int Win, int wShift, int hShift,
                    int stride, int pad)
{
  constexpr int K = KH * KW * 256;

  const int lane = threadIdx.x;
  const int mloc = lane & 15;
  const int g    = lane >> 4;

  // decompose row -> (n, ho, wo) with shifts (all dims power-of-two)
  const int row = blockIdx.x * 16 + mloc;
  const int wo  = row & ((1 << wShift) - 1);
  const int ho  = (row >> wShift) & ((1 << hShift) - 1);
  const int nb  = row >> (wShift + hShift);

  const int hbase = CONVT ? (ho - 2) : (ho * stride - pad);
  const int wbase = CONVT ? (wo - 2) : (wo * stride - pad);
  const long planeBase = (long)nb * Hin * Win;

  // 4 weight-column groups, 16 channels apart
  const int nb0 = blockIdx.y * 64 + mloc;
  const __bf16* wp0 = W + (size_t)nb0 * K + g * 8;

  v8f acc[4] = {v8f{}, v8f{}, v8f{}, v8f{}};

  for (int kh = 0; kh < KH; ++kh) {
    for (int kw = 0; kw < KW; ++kw) {
      int hi, wi;
      bool valid;
      if constexpr (!CONVT) {
        hi = hbase + kh;
        wi = wbase + kw;
        valid = ((unsigned)hi < (unsigned)Hin) & ((unsigned)wi < (unsigned)Win);
      } else {  // transposed conv k=4 s=2 p=1: tap valid iff parity matches
        const int h2 = hbase + kh, w2 = wbase + kw;
        hi = h2 >> 1; wi = w2 >> 1;
        valid = (!((h2 | w2) & 1)) &
                ((unsigned)hi < (unsigned)Hin) & ((unsigned)wi < (unsigned)Win);
      }
      if (!valid) continue;   // zero contribution -> skip whole tap

      const __bf16* ap = A + (((planeBase + (long)hi * Win + wi) << 8) + g * 8);
      const int kbase = (kh * KW + kw) << 8;

#pragma unroll
      for (int c = 0; c < 256; c += 32) {
        short8 lo  = load8(ap + c);        // K = kbase+c + g*8 .. +7
        short8 hi8 = load8(ap + c + 16);   // K = kbase+c + 16 + g*8 .. +7
        if constexpr (RELUIN) { lo = relu8(lo); hi8 = relu8(hi8); }
        BfPack a;
#pragma unroll
        for (int i = 0; i < 8; ++i) { a.s[i] = lo[i]; a.s[8 + i] = hi8[i]; }

#pragma unroll
        for (int j = 0; j < 4; ++j) {
          const __bf16* wpj = wp0 + (size_t)(16 * j) * K + kbase + c;
          short8 b0 = load8(wpj);          // K = kbase+c + g*8 .. +7
          short8 b1 = load8(wpj + 16);     // K = kbase+c + 16 + g*8 .. +7
          BfPack b;
#pragma unroll
          for (int i = 0; i < 8; ++i) { b.s[i] = b0[i]; b.s[8 + i] = b1[i]; }
          acc[j] = __builtin_amdgcn_wmma_f32_16x16x32_bf16(false, a.v, false, b.v,
                                                           (short)0, acc[j], false, false);
        }
      }
    }
  }

  // epilogue: D layout -> lane holds column ncol, rows g*8 + v
  const int mbase = blockIdx.x * 16 + g * 8;
  const int hwSh  = wShift + hShift;
#pragma unroll
  for (int j = 0; j < 4; ++j) {
    const int ncol = blockIdx.y * 64 + j * 16 + mloc;
    const float sc = scale[ncol];
    const float sh = shift[ncol];
#pragma unroll
    for (int v = 0; v < 8; ++v) {
      const int m = mbase + v;
      float val = acc[j][v];
      if constexpr (HASRES) val += (float)res[(size_t)m * N + ncol];
      val = val * sc + sh;
      if constexpr (RELUOUT) val = val > 0.f ? val : 0.f;
      if constexpr (HASOUTB) outB[(size_t)m * N + ncol] = (__bf16)val;
      if constexpr (F32MODE == 1) {          // row-major [M x N]
        outF[(size_t)m * N + ncol] = val;
      } else if constexpr (F32MODE == 2) {   // NCHW f32
        const int bi = m >> hwSh, hw = m & ((1 << hwSh) - 1);
        outF[(((size_t)bi * N + ncol) << hwSh) + hw] = val;
      }
    }
  }
}

// ------------------------- edge convs (C=3) -------------------------------
// conv1: 3->256 k4 s2 p1 + BN + ReLU, x f32 NCHW -> bf16 NHWC
__global__ void conv1_kernel(const float* __restrict__ x, const float* __restrict__ w,
                             const float* __restrict__ scale, const float* __restrict__ shift,
                             __bf16* __restrict__ out)
{
  const int idx = blockIdx.x * 256 + threadIdx.x;   // 16*64*64*256
  const int co = idx & 255;
  const int r  = idx >> 8;
  const int wo = r & 63, ho = (r >> 6) & 63, n = r >> 12;
  float acc = 0.f;
  for (int kh = 0; kh < 4; ++kh) {
    const int hi = ho * 2 - 1 + kh;
    if ((unsigned)hi >= 128u) continue;
    for (int kw = 0; kw < 4; ++kw) {
      const int wi = wo * 2 - 1 + kw;
      if ((unsigned)wi >= 128u) continue;
#pragma unroll
      for (int ci = 0; ci < 3; ++ci)
        acc += x[((n * 3 + ci) * 128 + hi) * 128 + wi] *
               w[((co * 3 + ci) * 4 + kh) * 4 + kw];
    }
  }
  float v = acc * scale[co] + shift[co];
  out[idx] = (__bf16)(v > 0.f ? v : 0.f);
}

// convT2: 256->3 k4 s2 p1 + tanh, bf16 NHWC -> f32 NCHW (d_out)
__global__ void convt2_kernel(const __bf16* __restrict__ A, const __bf16* __restrict__ W,
                              const float* __restrict__ shift, float* __restrict__ out)
{
  const int idx = blockIdx.x * 256 + threadIdx.x;   // 16*3*128*128
  const int wo = idx & 127, ho = (idx >> 7) & 127;
  const int t = idx >> 14;
  const int co = t % 3, n = t / 3;
  float acc = 0.f;
  for (int kh = 0; kh < 4; ++kh) {
    const int h2 = ho + kh - 2;
    if (h2 & 1) continue;
    const int hi = h2 >> 1;
    if ((unsigned)hi >= 64u) continue;
    for (int kw = 0; kw < 4; ++kw) {
      const int w2 = wo + kw - 2;
      if (w2 & 1) continue;
      const int wi = w2 >> 1;
      if ((unsigned)wi >= 64u) continue;
      const __bf16* ap = A + ((size_t)((n * 64 + hi) * 64 + wi) << 8);
      const __bf16* wp = W + (size_t)co * 4096 + ((kh * 4 + kw) << 8);
      for (int c = 0; c < 256; ++c) acc += (float)ap[c] * (float)wp[c];
    }
  }
  out[idx] = tanhf(acc + shift[co]);
}

// ---------------------- VQ argmax + codebook gather -----------------------
__global__ void vq_argmax(const float* __restrict__ S, const float* __restrict__ cb,
                          __bf16* __restrict__ zq, float* __restrict__ quantF)
{
  const int wave = (int)((blockIdx.x * blockDim.x + threadIdx.x) >> 5);
  const int lane = threadIdx.x & 31;
  if (wave >= 16384) return;
  const float* s = S + (size_t)wave * 512;
  float best = -3.4e38f; int bidx = 0x7fffffff;
  for (int j = lane; j < 512; j += 32) {
    const float v = s[j];
    if (v > best) { best = v; bidx = j; }
  }
#pragma unroll
  for (int off = 16; off; off >>= 1) {
    const float ob = __shfl_xor(best, off, 32);
    const int   oi = __shfl_xor(bidx, off, 32);
    if (ob > best || (ob == best && oi < bidx)) { best = ob; bidx = oi; }
  }
  const int b = wave >> 10, hw = wave & 1023;
  const float* c = cb + (size_t)bidx * 256;
  for (int ch = lane; ch < 256; ch += 32) {
    const float q = c[ch];
    zq[(size_t)wave * 256 + ch] = (__bf16)q;                    // decoder input (== z_q)
    quantF[((size_t)b * 256 + ch) * 1024 + hw] = q;             // quantized, NCHW f32
  }
}

// ---------------------------- prep kernels --------------------------------
__global__ void prep_ss(float* __restrict__ scale, float* __restrict__ shift,
                        const float* __restrict__ bias,
                        const float* __restrict__ g, const float* __restrict__ b,
                        const float* __restrict__ m, const float* __restrict__ v, int C)
{
  const int i = blockIdx.x * blockDim.x + threadIdx.x;
  if (i >= C) return;
  const float bi = bias ? bias[i] : 0.f;
  if (g) {
    const float s = g[i] * rsqrtf(v[i] + 1e-5f);
    scale[i] = s;
    shift[i] = (bi - m[i]) * s + b[i];
  } else {
    scale[i] = 1.f;
    shift[i] = bi;
  }
}

__global__ void prep_code(const float* __restrict__ cb,
                          float* __restrict__ scale, float* __restrict__ shift)
{
  const int n = blockIdx.x * blockDim.x + threadIdx.x;
  if (n >= 512) return;
  float s = 0.f;
  for (int k = 0; k < 256; ++k) { const float c = cb[n * 256 + k]; s += c * c; }
  scale[n] = 1.f;
  shift[n] = -0.5f * s;
}

// conv weight repack: src f32 [Co][Ci][Kh][Kw] -> dst bf16 [Co][Kh][Kw][Ci]
__global__ void repack_conv_w(const float* __restrict__ src, __bf16* __restrict__ dst,
                              int Co, int Ci, int Kh, int Kw)
{
  const int idx = blockIdx.x * 256 + threadIdx.x;
  const int total = Co * Ci * Kh * Kw;
  if (idx >= total) return;
  const int ci = idx % Ci; int t = idx / Ci;
  const int kw = t % Kw; t /= Kw;
  const int kh = t % Kh;
  const int co = t / Kh;
  dst[idx] = (__bf16)src[((co * Ci + ci) * Kh + kh) * Kw + kw];
}

// convT weight repack: src f32 [Ci][Co][Kh][Kw] -> dst bf16 [Co][Kh][Kw][Ci], flipped
__global__ void repack_convt_w(const float* __restrict__ src, __bf16* __restrict__ dst,
                               int Co, int Ci, int Kh, int Kw)
{
  const int idx = blockIdx.x * 256 + threadIdx.x;
  const int total = Co * Ci * Kh * Kw;
  if (idx >= total) return;
  const int ci = idx % Ci; int t = idx / Ci;
  const int kw = t % Kw; t /= Kw;
  const int kh = t % Kh;
  const int co = t / Kh;
  dst[idx] = (__bf16)src[((ci * Co + co) * Kh + (Kh - 1 - kh)) * Kw + (Kw - 1 - kw)];
}

__global__ void cvt_bf16(const float* __restrict__ src, __bf16* __restrict__ dst, int n)
{
  const int i = blockIdx.x * 256 + threadIdx.x;
  if (i < n) dst[i] = (__bf16)src[i];
}

// ==========================================================================
// host side
// ==========================================================================
enum {
  IN_X = 0, IN_EC1_W = 1, IN_EC1_B = 2, IN_EBN1 = 3,
  IN_EC2_W = 7, IN_EC2_B = 8, IN_EBN2 = 9,
  IN_RB1C1_W = 13, IN_RB1C1_B = 14, IN_RB1BN = 15,
  IN_RB1C2_W = 19, IN_RB1C2_B = 20, IN_EBN3 = 21,
  IN_RB2C1_W = 25, IN_RB2C1_B = 26, IN_RB2BN = 27,
  IN_RB2C2_W = 31, IN_RB2C2_B = 32, IN_EBN4 = 33,
  IN_RB3C1_W = 37, IN_RB3C1_B = 38, IN_RB3C2_W = 39, IN_RB3C2_B = 40, IN_DBN1 = 41,
  IN_RB4C1_W = 45, IN_RB4C1_B = 46, IN_RB4C2_W = 47, IN_RB4C2_B = 48,
  IN_DCT1_W = 49, IN_DCT1_B = 50, IN_DBN2 = 51,
  IN_DCT2_W = 55, IN_DCT2_B = 56, IN_CB = 57
};

extern "C" void kernel_launch(void* const* d_in, const int* in_sizes, int n_in,
                              void* d_out, int out_size, void* d_ws, size_t ws_size,
                              hipStream_t stream)
{
  (void)in_sizes; (void)n_in; (void)out_size; (void)ws_size;

  char* ws = (char*)d_ws;
  size_t off = 0;
  auto alloc = [&](size_t bytes) -> char* {
    char* p = ws + off;
    off += (bytes + 255) & ~(size_t)255;
    return p;
  };

  // ---- weights (bf16, [Co][K]) ----
  __bf16* wec2   = (__bf16*)alloc((size_t)256 * 4096 * 2);
  __bf16* wrb1c1 = (__bf16*)alloc((size_t)256 * 2304 * 2);
  __bf16* wrb1c2 = (__bf16*)alloc((size_t)256 * 256 * 2);
  __bf16* wrb2c1 = (__bf16*)alloc((size_t)256 * 2304 * 2);
  __bf16* wrb2c2 = (__bf16*)alloc((size_t)256 * 256 * 2);
  __bf16* wrb3c1 = (__bf16*)alloc((size_t)256 * 2304 * 2);
  __bf16* wrb3c2 = (__bf16*)alloc((size_t)256 * 256 * 2);
  __bf16* wrb4c1 = (__bf16*)alloc((size_t)256 * 2304 * 2);
  __bf16* wrb4c2 = (__bf16*)alloc((size_t)256 * 256 * 2);
  __bf16* wdct1  = (__bf16*)alloc((size_t)256 * 4096 * 2);
  __bf16* wdct2  = (__bf16*)alloc((size_t)3 * 4096 * 2);
  __bf16* wcb    = (__bf16*)alloc((size_t)512 * 256 * 2);

  // ---- scale/shift pairs (13 x 1024 f32) ----
  float* ssb = (float*)alloc(13 * 1024 * sizeof(float));
  auto SC = [&](int i) { return ssb + (size_t)i * 1024; };
  auto SH = [&](int i) { return ssb + (size_t)i * 1024 + 512; };
  enum { P_EC1, P_EC2, P_RB1, P_EBN3, P_RB2, P_EBN4, P_RB3C1, P_DBN1,
         P_RB4C1, P_RB4C2, P_DCT1, P_DCT2, P_CODE };

  // ---- activations ----
  __bf16* bufH1 = (__bf16*)alloc((size_t)16 * 64 * 64 * 256 * 2); // 33.5MB, reused for D3
  __bf16* bufA  = (__bf16*)alloc((size_t)16 * 32 * 32 * 256 * 2); // A2, later Zq
  __bf16* bufB  = (__bf16*)alloc((size_t)16 * 32 * 32 * 256 * 2); // resblock temp
  __bf16* bufC  = (__bf16*)alloc((size_t)16 * 32 * 32 * 256 * 2); // A3, later D1
  __bf16* bufD  = (__bf16*)alloc((size_t)16 * 32 * 32 * 256 * 2); // z_e bf16, later D2
  float*  bufS  = (float*) alloc((size_t)16384 * 512 * 4);        // VQ scores

  float* outAll  = (float*)d_out;
  float* decoded = outAll;                       // 16*3*128*128
  float* z_e     = outAll + 786432;              // 16*256*32*32
  float* quant   = z_e + 4194304;                // 16*256*32*32

  const float* F = nullptr;
  auto in = [&](int i) { return (const float*)d_in[i]; };

  // -------------------- prep: weight repack + scale/shift --------------------
  auto rp  = [&](const float* s, __bf16* d, int Co, int Ci, int Kh, int Kw) {
    int tot = Co * Ci * Kh * Kw;
    repack_conv_w<<<(tot + 255) / 256, 256, 0, stream>>>(s, d, Co, Ci, Kh, Kw);
  };
  auto rpt = [&](const float* s, __bf16* d, int Co, int Ci, int Kh, int Kw) {
    int tot = Co * Ci * Kh * Kw;
    repack_convt_w<<<(tot + 255) / 256, 256, 0, stream>>>(s, d, Co, Ci, Kh, Kw);
  };
  rp(in(IN_EC2_W),  wec2,   256, 256, 4, 4);
  rp(in(IN_RB1C1_W), wrb1c1, 256, 256, 3, 3);
  rp(in(IN_RB1C2_W), wrb1c2, 256, 256, 1, 1);
  rp(in(IN_RB2C1_W), wrb2c1, 256, 256, 3, 3);
  rp(in(IN_RB2C2_W), wrb2c2, 256, 256, 1, 1);
  rp(in(IN_RB3C1_W), wrb3c1, 256, 256, 3, 3);
  rp(in(IN_RB3C2_W), wrb3c2, 256, 256, 1, 1);
  rp(in(IN_RB4C1_W), wrb4c1, 256, 256, 3, 3);
  rp(in(IN_RB4C2_W), wrb4c2, 256, 256, 1, 1);
  rpt(in(IN_DCT1_W), wdct1, 256, 256, 4, 4);
  rpt(in(IN_DCT2_W), wdct2, 3,   256, 4, 4);
  cvt_bf16<<<(131072 + 255) / 256, 256, 0, stream>>>(in(IN_CB), wcb, 131072);

  auto ss = [&](int pair, int biasIdx, int bnIdx, int C) {
    const float* g = bnIdx >= 0 ? in(bnIdx) : F;
    const float* b = bnIdx >= 0 ? in(bnIdx + 1) : F;
    const float* m = bnIdx >= 0 ? in(bnIdx + 2) : F;
    const float* v = bnIdx >= 0 ? in(bnIdx + 3) : F;
    prep_ss<<<(C + 255) / 256, 256, 0, stream>>>(SC(pair), SH(pair),
                                                 in(biasIdx), g, b, m, v, C);
  };
  ss(P_EC1,   IN_EC1_B,   IN_EBN1,  256);
  ss(P_EC2,   IN_EC2_B,   IN_EBN2,  256);
  ss(P_RB1,   IN_RB1C1_B, IN_RB1BN, 256);
  ss(P_EBN3,  IN_RB1C2_B, IN_EBN3,  256);
  ss(P_RB2,   IN_RB2C1_B, IN_RB2BN, 256);
  ss(P_EBN4,  IN_RB2C2_B, IN_EBN4,  256);
  ss(P_RB3C1, IN_RB3C1_B, -1,       256);
  ss(P_DBN1,  IN_RB3C2_B, IN_DBN1,  256);
  ss(P_RB4C1, IN_RB4C1_B, -1,       256);
  ss(P_RB4C2, IN_RB4C2_B, -1,       256);
  ss(P_DCT1,  IN_DCT1_B,  IN_DBN2,  256);
  ss(P_DCT2,  IN_DCT2_B,  -1,       3);
  prep_code<<<2, 256, 0, stream>>>(in(IN_CB), SC(P_CODE), SH(P_CODE));

  // ------------------------------- pipeline ---------------------------------
  const dim3 blk(32);
  const dim3 g32(16384 / 16, 256 / 64);     // M=16384, N=256
  const dim3 gvq(16384 / 16, 512 / 64);     // M=16384, N=512
  const dim3 gct(65536 / 16, 256 / 64);     // M=65536, N=256

  // encoder conv1 (3->256) + ebn1 + relu          -> bufH1 [16,64,64,256]
  conv1_kernel<<<65536, 256, 0, stream>>>(in(IN_X), in(IN_EC1_W),
                                          SC(P_EC1), SH(P_EC1), bufH1);
  // encoder conv2 (4x4 s2) + ebn2 + relu          -> bufA  [16,32,32,256]
  wmma_conv_gemm<4,4,0,0,1,0,0,1><<<g32, blk, 0, stream>>>(
      bufH1, wec2, bufA, nullptr, nullptr, SC(P_EC2), SH(P_EC2),
      256, 64, 64, 5, 5, 2, 1);
  // rb1: 3x3(relu in) + rb1_bn                    -> bufB
  wmma_conv_gemm<3,3,0,1,0,0,0,1><<<g32, blk, 0, stream>>>(
      bufA, wrb1c1, bufB, nullptr, nullptr, SC(P_RB1), SH(P_RB1),
      256, 32, 32, 5, 5, 1, 1);
  // rb1: 1x1(relu in) + res(bufA) + ebn3          -> bufC
  wmma_conv_gemm<1,1,0,1,0,0,1,1><<<g32, blk, 0, stream>>>(
      bufB, wrb1c2, bufC, nullptr, bufA, SC(P_EBN3), SH(P_EBN3),
      256, 32, 32, 5, 5, 1, 0);
  // rb2: 3x3(relu in) + rb2_bn                    -> bufB
  wmma_conv_gemm<3,3,0,1,0,0,0,1><<<g32, blk, 0, stream>>>(
      bufC, wrb2c1, bufB, nullptr, nullptr, SC(P_RB2), SH(P_RB2),
      256, 32, 32, 5, 5, 1, 1);
  // rb2: 1x1(relu in) + res(bufC) + ebn4 = z_e    -> bufD (bf16) + z_e (f32 NCHW)
  wmma_conv_gemm<1,1,0,1,0,2,1,1><<<g32, blk, 0, stream>>>(
      bufB, wrb2c2, bufD, z_e, bufC, SC(P_EBN4), SH(P_EBN4),
      256, 32, 32, 5, 5, 1, 0);
  // VQ scores: z.c - 0.5||c||^2                   -> bufS [16384 x 512]
  wmma_conv_gemm<1,1,0,0,0,1,0,0><<<gvq, blk, 0, stream>>>(
      bufD, wcb, nullptr, bufS, nullptr, SC(P_CODE), SH(P_CODE),
      512, 32, 32, 5, 5, 1, 0);
  // argmax + gather: z_q = q                      -> bufA (bf16) + quant (f32 NCHW)
  vq_argmax<<<2048, 256, 0, stream>>>(bufS, in(IN_CB), bufA, quant);
  // rb3: 3x3(relu in) + bias                      -> bufB
  wmma_conv_gemm<3,3,0,1,0,0,0,1><<<g32, blk, 0, stream>>>(
      bufA, wrb3c1, bufB, nullptr, nullptr, SC(P_RB3C1), SH(P_RB3C1),
      256, 32, 32, 5, 5, 1, 1);
  // rb3: 1x1(relu in) + res(bufA) + dbn1          -> bufC
  wmma_conv_gemm<1,1,0,1,0,0,1,1><<<g32, blk, 0, stream>>>(
      bufB, wrb3c2, bufC, nullptr, bufA, SC(P_DBN1), SH(P_DBN1),
      256, 32, 32, 5, 5, 1, 0);
  // rb4: 3x3(relu in) + bias                      -> bufB
  wmma_conv_gemm<3,3,0,1,0,0,0,1><<<g32, blk, 0, stream>>>(
      bufC, wrb4c1, bufB, nullptr, nullptr, SC(P_RB4C1), SH(P_RB4C1),
      256, 32, 32, 5, 5, 1, 1);
  // rb4: 1x1(relu in) + res(bufC) + bias          -> bufD
  wmma_conv_gemm<1,1,0,1,0,0,1,1><<<g32, blk, 0, stream>>>(
      bufB, wrb4c2, bufD, nullptr, bufC, SC(P_RB4C2), SH(P_RB4C2),
      256, 32, 32, 5, 5, 1, 0);
  // convT1 (256->256, 2x up) + dbn2 + relu        -> bufH1 [16,64,64,256]
  wmma_conv_gemm<4,4,1,0,1,0,0,1><<<gct, blk, 0, stream>>>(
      bufD, wdct1, bufH1, nullptr, nullptr, SC(P_DCT1), SH(P_DCT1),
      256, 32, 32, 6, 6, 1, 0);
  // convT2 (256->3, 2x up) + tanh                 -> decoded (f32 NCHW)
  convt2_kernel<<<3072, 256, 0, stream>>>(bufH1, wdct2, SH(P_DCT2), decoded);
}
